// MMHSE_local_69870527971927
// MI455X (gfx1250) — matrix operand order repaired and checked by reading, not compiled, not run on hardware
//
#include <hip/hip_runtime.h>
#include <hip/hip_bf16.h>

typedef __bf16 bf16;
typedef __attribute__((ext_vector_type(16))) __bf16 v16bf;
typedef __attribute__((ext_vector_type(8)))  float  v8f;

namespace {
constexpr float kEPS    = 1e-8f;
constexpr float kLEAKY  = 0.1f;
constexpr float kSMOOTH = 9.0f;
constexpr float kNEG    = -1e30f;

constexpr int BATCH = 128;
constexpr int RREG  = 36;     // cnn regions
constexpr int CNND  = 2048;
constexpr int NPAT  = 9;      // bovw patches
constexpr int EMBD  = 1024;
constexpr int WORDD = 300;
constexpr int WPADD = 320;    // word dim padded to mult of 64
constexpr int MAXL  = 40;
constexpr int RP    = 48;     // padded regions/words (mult of 16)
constexpr int PP    = 16;     // padded patches
}

// ---------------- WMMA helpers (CDNA5 bf16 16x16x32, f32 accum) ----------------

__device__ inline v8f wmma_bf16(v16bf a, v16bf b, v8f c) {
  return __builtin_amdgcn_wmma_f32_16x16x32_bf16(false, a, false, b, (short)0, c,
                                                 false, false);
}

// A fragment: 16 rows (M) x 32 K, row-major source, ld = row stride (elements).
// Contiguous K order used here (same as load_frag_n): lanes 0-15 M=lane K=0..15; lanes 16-31 M=lane-16 K=16..31.
__device__ inline v16bf load_frag_m(const bf16* __restrict__ base, int ld) {
  const int lane = threadIdx.x & 31;
  const int g = lane >> 4;
  const bf16* p = base + (long)(lane & 15) * ld;
  v16bf a;
#pragma unroll
  for (int pr = 0; pr < 8; ++pr) {
    const int k = 16 * g + 2 * pr;
    a[2 * pr]     = p[k];
    a[2 * pr + 1] = p[k + 1];
  }
  return a;
}

// B fragment: 32 K x 16 cols (N); source is N-major (row n holds K contiguous), ld = row stride.
// lanes 0-15: N=lane, K=0..15; lanes 16-31: N=lane-16, K=16..31.
__device__ inline v16bf load_frag_n(const bf16* __restrict__ base, int ld) {
  const int lane = threadIdx.x & 31;
  const bf16* p = base + (long)(lane & 15) * ld + ((lane >> 4) * 16);
  v16bf b;
#pragma unroll
  for (int e = 0; e < 16; ++e) b[e] = p[e];
  return b;
}

// Cooperative stage of `nrows` rows x W K-elements of an N-major bf16 matrix into an
// LDS panel laid out [row][W]. Copies as u32 (2 bf16), coalesced across threads.
template <int W>
__device__ inline void stage_rows(bf16* __restrict__ dst, const bf16* __restrict__ src,
                                  long ld, int nrows, int tid, int nthr) {
  constexpr int UPR = W / 2;  // u32 per row
  const int total = nrows * UPR;
  for (int i = tid; i < total; i += nthr) {
    const int r = i / UPR, c = i % UPR;
    ((uint32_t*)dst)[i] = *(const uint32_t*)(src + (long)r * ld + 2 * c);
  }
}

__device__ inline float wred_sum(float v) {
#pragma unroll
  for (int o = 16; o > 0; o >>= 1) v += __shfl_xor(v, o, 32);
  return v;
}
__device__ inline float wred_max(float v) {
#pragma unroll
  for (int o = 16; o > 0; o >>= 1) v = fmaxf(v, __shfl_xor(v, o, 32));
  return v;
}

// ---------------- prep kernels ----------------

__global__ void k_cvt(const float* __restrict__ src, bf16* __restrict__ dst, long n) {
  long i = (long)blockIdx.x * 256 + threadIdx.x;
  if (i < n) dst[i] = (bf16)src[i];
}

__global__ void k_zero(float* __restrict__ p, long n) {
  long i = (long)blockIdx.x * 256 + threadIdx.x;
  if (i < n) p[i] = 0.f;
}

// dst[n*K+k] = src[k*N+n]
__global__ void k_transpose(const float* __restrict__ src, bf16* __restrict__ dst,
                            int K, int N) {
  long i = (long)blockIdx.x * 256 + threadIdx.x;
  if (i >= (long)K * N) return;
  int n = (int)(i / K), k = (int)(i % K);
  dst[i] = (bf16)src[(long)k * N + n];
}

// dst rows x dcols, src rows x scols, zero-pad columns >= scols
__global__ void k_padcvt(const float* __restrict__ src, bf16* __restrict__ dst,
                         int rows, int scols, int dcols) {
  long i = (long)blockIdx.x * 256 + threadIdx.x;
  if (i >= (long)rows * dcols) return;
  int r = (int)(i / dcols), c = (int)(i % dcols);
  dst[i] = (c < scols) ? (bf16)src[(long)r * scols + c] : (bf16)0.f;
}

// x_pad[b][l][k] = emb[captions[b][l]][k], zero-padded to WPADD
__global__ void k_gather(const int* __restrict__ captions,
                         const float* __restrict__ table, bf16* __restrict__ dst) {
  long i = (long)blockIdx.x * 256 + threadIdx.x;
  if (i >= (long)BATCH * MAXL * WPADD) return;
  int k = (int)(i % WPADD);
  long bl = i / WPADD;
  int tok = captions[bl];
  dst[i] = (k < WORDD) ? (bf16)table[(long)tok * WORDD + k] : (bf16)0.f;
}

// ---------------- GEMM: C(MxN,f32) = A(MxK,bf16) * Bt(N-major,bf16) + bias ----------------
// block 256 = 8 waves; wave = 16x16 tile; block tile 128 rows x 16 cols.
// B panel (shared by all 8 waves) staged through LDS, 64-wide K chunks.
__global__ void k_gemm_bias(const bf16* __restrict__ A, const bf16* __restrict__ Bt,
                            const float* __restrict__ bias, float* __restrict__ C,
                            int M, int N, int K) {
  __shared__ bf16 sB[16 * 64];
  const int tid = threadIdx.x;
  const int w = tid >> 5;
  const int m0 = blockIdx.y * 128 + w * 16;
  const int n0 = blockIdx.x * 16;
  v8f acc = {};
  for (int k = 0; k < K; k += 64) {
    if (k + 64 < K)  // warm next A chunk (global_prefetch)
      __builtin_prefetch(A + (long)(m0 + (tid & 15)) * K + k + 64, 0, 0);
    __syncthreads();
    stage_rows<64>(sB, Bt + (long)n0 * K + k, K, 16, tid, 256);
    __syncthreads();
#pragma unroll
    for (int kk = 0; kk < 64; kk += 32) {
      v16bf fa = load_frag_m(A + (long)m0 * K + k + kk, K);
      v16bf fb = load_frag_n(sB + kk, 64);
      acc = wmma_bf16(fa, fb, acc);
    }
  }
  const int lane = tid & 31, g = lane >> 4, n = lane & 15;
  const float bv = bias ? bias[n0 + n] : 0.f;
#pragma unroll
  for (int v = 0; v < 8; ++v) {
    const int m = m0 + v + 8 * g;
    C[(long)m * N + n0 + n] = acc[v] + bv;
  }
}

// ---------------- row l2norm finalize: v = scale*(in0[+in1]); out=v/sqrt(sum((v+eps)^2)) ----------------
__global__ void k_finalize(const float* __restrict__ in0, const float* __restrict__ in1,
                           float scale, bf16* __restrict__ out, float* __restrict__ norms,
                           int rpsIn, int rpsOut, int d) {
  const int q = blockIdx.x, s = blockIdx.y;
  const long orow = ((long)s * rpsOut + q) * d;
  if (q >= rpsIn) {
    for (int i = threadIdx.x; i < d; i += 256) out[orow + i] = (bf16)0.f;
    if (threadIdx.x == 0) norms[(long)s * rpsOut + q] = 0.f;
    return;
  }
  const long irow = ((long)s * rpsIn + q) * d;
  float s1 = 0.f, s2 = 0.f;
  for (int i = threadIdx.x; i < d; i += 256) {
    float v = in0[irow + i];
    if (in1) v += in1[irow + i];
    v *= scale;
    const float e = v + kEPS;
    s1 += e * e;
    s2 += v * v;
  }
  __shared__ float sh1[8], sh2[8];
  s1 = wred_sum(s1);
  s2 = wred_sum(s2);
  if ((threadIdx.x & 31) == 0) { sh1[threadIdx.x >> 5] = s1; sh2[threadIdx.x >> 5] = s2; }
  __syncthreads();
  if (threadIdx.x == 0) {
    float t1 = 0.f, t2 = 0.f;
    for (int i = 0; i < 8; ++i) { t1 += sh1[i]; t2 += sh2[i]; }
    const float inv = rsqrtf(t1);
    sh1[0] = inv;
    norms[(long)s * rpsOut + q] = sqrtf(t2) * inv;
  }
  __syncthreads();
  const float inv = sh1[0];
  for (int i = threadIdx.x; i < d; i += 256) {
    float v = in0[irow + i];
    if (in1) v += in1[irow + i];
    out[orow + i] = (bf16)(v * scale * inv);
  }
}

// ---------------- Gram: G_s = E_s * E_s^T (RPs x RPs), per sample ----------------
// Single LDS panel feeds both A- and B-fragments (rows are the same matrix).
__global__ __launch_bounds__(288) void k_gram(const bf16* __restrict__ E,
                                              float* __restrict__ G, int RPs, int d) {
  __shared__ bf16 sP[48 * 64];
  const int s = blockIdx.x;
  const bf16* Es = E + (long)s * RPs * d;
  float* Gs = G + (long)s * RPs * RPs;
  const int tb = RPs >> 4;
  const int tid = threadIdx.x;
  const int w = tid >> 5;
  const int ta = w / tb, tc = w % tb;
  const bool own = w < tb * tb;
  v8f acc = {};
  for (int k = 0; k < d; k += 64) {
    __syncthreads();
    stage_rows<64>(sP, Es + k, d, RPs, tid, 288);
    __syncthreads();
    if (own) {
#pragma unroll
      for (int kk = 0; kk < 64; kk += 32) {
        v16bf fa = load_frag_m(sP + (long)(ta * 16) * 64 + kk, 64);
        v16bf fb = load_frag_n(sP + (long)(tc * 16) * 64 + kk, 64);
        acc = wmma_bf16(fa, fb, acc);
      }
    }
  }
  if (own) {
    const int lane = tid & 31, g = lane >> 4, n = lane & 15;
#pragma unroll
    for (int v = 0; v < 8; ++v)
      Gs[(long)(ta * 16 + v + 8 * g) * RPs + tc * 16 + n] = acc[v];
  }
}

// ---------------- fused GRU step (4 dir-streams, one time step) ----------------
// grid=(64,1,4), block 256. Stream s: model=s>>1, dir=s&1. Column block of 16 hidden cols,
// all 128 batch rows. B panels (3 gates, shared by all 8 waves) staged through LDS.
// Accumulators: r,z combined (x*Wih + h*Whh), xn and hn separate (n = tanh(xn + r*hn)).
__global__ void k_gru_step(const bf16* __restrict__ x_pad,   // [2][128][40][320]
                           const bf16* __restrict__ wih_all, // [4][3072][320] N-major
                           const bf16* __restrict__ whh_all, // [4][3072][1024] N-major
                           const float* __restrict__ c_bih, const float* __restrict__ c_bhh,
                           const float* __restrict__ b_bih, const float* __restrict__ b_bhh,
                           const int* __restrict__ lengths,
                           const float* __restrict__ h_in_f32, const bf16* __restrict__ h_in_bf,
                           float* __restrict__ h_out_f32, bf16* __restrict__ h_out_bf,
                           float* __restrict__ out_all,      // [4][128][40][1024]
                           int t) {
  __shared__ bf16 sB[3 * 16 * 64];
  const int s = blockIdx.z;
  const int model = s >> 1, dir = s & 1;
  const int lpos = dir ? (MAXL - 1 - t) : t;
  const int tid = threadIdx.x;
  const int w = tid >> 5;
  const int m0 = w * 16;            // batch rows
  const int n0 = blockIdx.x * 16;   // hidden column block

  const bf16* hs   = h_in_bf + (long)s * BATCH * EMBD;
  const bf16* whhs = whh_all + (long)s * 3 * EMBD * EMBD;
  const bf16* wihs = wih_all + (long)s * 3 * EMBD * WPADD;
  const bf16* xs   = x_pad + ((long)model * BATCH * MAXL + lpos) * WPADD;

  v8f ar = {}, az = {}, axn = {}, ahn = {};
  for (int k = 0; k < EMBD; k += 64) {           // h @ Whh^T (r, z, hn)
    __syncthreads();
    stage_rows<64>(sB,            whhs + (long)n0 * EMBD + k,              EMBD, 16, tid, 256);
    stage_rows<64>(sB + 16 * 64,  whhs + (long)(EMBD + n0) * EMBD + k,     EMBD, 16, tid, 256);
    stage_rows<64>(sB + 32 * 64,  whhs + (long)(2 * EMBD + n0) * EMBD + k, EMBD, 16, tid, 256);
    __syncthreads();
#pragma unroll
    for (int kk = 0; kk < 64; kk += 32) {
      v16bf fa = load_frag_m(hs + (long)m0 * EMBD + k + kk, EMBD);
      ar  = wmma_bf16(fa, load_frag_n(sB + kk, 64), ar);
      az  = wmma_bf16(fa, load_frag_n(sB + 16 * 64 + kk, 64), az);
      ahn = wmma_bf16(fa, load_frag_n(sB + 32 * 64 + kk, 64), ahn);
    }
  }
  for (int k = 0; k < WPADD; k += 64) {          // x @ Wih^T (r, z, xn)
    __syncthreads();
    stage_rows<64>(sB,            wihs + (long)n0 * WPADD + k,              WPADD, 16, tid, 256);
    stage_rows<64>(sB + 16 * 64,  wihs + (long)(EMBD + n0) * WPADD + k,     WPADD, 16, tid, 256);
    stage_rows<64>(sB + 32 * 64,  wihs + (long)(2 * EMBD + n0) * WPADD + k, WPADD, 16, tid, 256);
    __syncthreads();
#pragma unroll
    for (int kk = 0; kk < 64; kk += 32) {
      v16bf fa = load_frag_m(xs + (long)m0 * (MAXL * WPADD) + k + kk, MAXL * WPADD);
      ar  = wmma_bf16(fa, load_frag_n(sB + kk, 64), ar);
      az  = wmma_bf16(fa, load_frag_n(sB + 16 * 64 + kk, 64), az);
      axn = wmma_bf16(fa, load_frag_n(sB + 32 * 64 + kk, 64), axn);
    }
  }

  const float* bih = (model ? b_bih : c_bih) + (long)dir * 3 * EMBD;
  const float* bhh = (model ? b_bhh : c_bhh) + (long)dir * 3 * EMBD;
  const int lane = tid & 31, g = lane >> 4, n = lane & 15;
  const int col = n0 + n;
  const float br_ = bih[col] + bhh[col];
  const float bz_ = bih[EMBD + col] + bhh[EMBD + col];
  const float bxn = bih[2 * EMBD + col];
  const float bhn = bhh[2 * EMBD + col];

  const float* hfs = h_in_f32 + (long)s * BATCH * EMBD;
  float* hof = h_out_f32 + (long)s * BATCH * EMBD;
  bf16*  hob = h_out_bf + (long)s * BATCH * EMBD;
  float* outs = out_all + (long)s * BATCH * MAXL * EMBD;

#pragma unroll
  for (int v = 0; v < 8; ++v) {
    const int b = m0 + v + 8 * g;
    const float r = 1.f / (1.f + expf(-(ar[v] + br_)));
    const float z = 1.f / (1.f + expf(-(az[v] + bz_)));
    const float nn = tanhf(axn[v] + bxn + r * (ahn[v] + bhn));
    const float hp = hfs[(long)b * EMBD + col];
    const float m = (lpos < lengths[b]) ? 1.f : 0.f;
    const float hnew = (m > 0.f) ? ((1.f - z) * nn + z * hp) : hp;
    hof[(long)b * EMBD + col] = hnew;
    hob[(long)b * EMBD + col] = (bf16)hnew;
    outs[((long)b * MAXL + lpos) * EMBD + col] = hnew * m;
  }
}

// ---------------- fused pair cross-attention similarity ----------------
// One directional SCAN similarity: ctx index x (count Xc), query index y (count Yc).
// S in LDS holds raw dots S[a*49+b] (a = A rows, b = B rows). ctxIsA selects orientation.
// Gram trick: |wctx|^2 = w^T G_ctx w.
__device__ float dir_sim(const float* __restrict__ S, int ctxIsA, int Xc, int Yc,
                         const float* __restrict__ G, int gs,
                         const float* __restrict__ qn,
                         const float* ymask, const float* xmask, float finScale,
                         float* rn, float (*wrow)[RP], float* red) {
  const int tid = threadIdx.x;
  const int wv = tid >> 5, ln = tid & 31;
  const int NW = 9;

  // per-context clipped-l2 normalizer over (masked) queries
  for (int x = tid; x < Xc; x += 288) {
    float sacc = 0.f;
    for (int y = 0; y < Yc; ++y) {
      float v = ctxIsA ? S[x * 49 + y] : S[y * 49 + x];
      v = v >= 0.f ? v : kLEAKY * v;
      const float e = v + kEPS;
      sacc += e * e * (ymask ? ymask[y] : 1.f);
    }
    rn[x] = rsqrtf(sacc);
  }
  __syncthreads();

  float acc = 0.f;
  for (int y = wv; y < Yc; y += NW) {
    const int x0 = ln, x1 = ln + 32;
    float raw0 = 0.f, raw1 = 0.f, t0 = -3.0e38f, t1 = -3.0e38f;
    if (x0 < Xc) {
      raw0 = ctxIsA ? S[x0 * 49 + y] : S[y * 49 + x0];
      float v = raw0 >= 0.f ? raw0 : kLEAKY * raw0;
      t0 = kSMOOTH * v * rn[x0];
      if (xmask && xmask[x0] == 0.f) t0 = kNEG;
    }
    if (x1 < Xc) {
      raw1 = ctxIsA ? S[x1 * 49 + y] : S[y * 49 + x1];
      float v = raw1 >= 0.f ? raw1 : kLEAKY * raw1;
      t1 = kSMOOTH * v * rn[x1];
      if (xmask && xmask[x1] == 0.f) t1 = kNEG;
    }
    const float mx = wred_max(fmaxf(t0, t1));
    const float e0 = (x0 < Xc) ? expf(t0 - mx) : 0.f;
    const float e1 = (x1 < Xc) ? expf(t1 - mx) : 0.f;
    const float inv = 1.f / wred_sum(e0 + e1);
    const float w0 = e0 * inv, w1 = e1 * inv;
    wrow[wv][ln] = w0;
    if (ln < RP - 32) wrow[wv][ln + 32] = w1;
    __builtin_amdgcn_wave_barrier();

    const float nump = wred_sum(w0 * raw0 + w1 * raw1);   // sum_x w * rawS

    float qp = 0.f;                                       // w^T G w
    if (x0 < Xc) {
      float inner = 0.f;
      for (int xp = 0; xp < Xc; ++xp) inner += G[(long)x0 * gs + xp] * wrow[wv][xp];
      qp += w0 * inner;
    }
    if (x1 < Xc) {
      float inner = 0.f;
      for (int xp = 0; xp < Xc; ++xp) inner += G[(long)x1 * gs + xp] * wrow[wv][xp];
      qp += w1 * inner;
    }
    qp = wred_sum(qp);
    __builtin_amdgcn_wave_barrier();

    if (ln == 0) {
      const float den = fmaxf(qn[y] * sqrtf(fmaxf(qp, 0.f)), kEPS);
      acc += (nump / den) * (ymask ? ymask[y] : 1.f);
    }
  }
  if (ln == 0) red[wv] = acc;
  __syncthreads();
  float r = 0.f;
  if (tid == 0) {
    for (int i = 0; i < NW; ++i) r += red[i];
    r *= finScale;
  }
  __syncthreads();
  return r;
}

// grid=(128,128), block 288 (9 waves). Computes both attention directions for one sample pair.
// A/B K-panels staged through LDS (shared by all tile waves).
__global__ __launch_bounds__(288) void k_pair(
    const bf16* __restrict__ Aall, const bf16* __restrict__ Ball,
    const float* __restrict__ GAall, const float* __restrict__ GBall,
    const float* __restrict__ nAall, const float* __restrict__ nBall,
    int Ra, int Rb, int RaP, int RbP,
    const int* lengths,   // non-null => B rows are words (masked); t2i-style dir gets /len
    float* __restrict__ out) {
  __shared__ float S[48 * 49];
  __shared__ float GA[48 * 48];
  __shared__ float GB[48 * 48];
  __shared__ float nA[48], nB[48], msk[48];
  __shared__ float rn[48];
  __shared__ float wrow[9][RP];
  __shared__ float red[9];
  __shared__ bf16 sA[48 * 64];
  __shared__ bf16 sB[48 * 64];

  const int ib = blockIdx.x, ia = blockIdx.y;
  const int tid = threadIdx.x;
  const bf16* A = Aall + (long)ia * RaP * EMBD;
  const bf16* B = Ball + (long)ib * RbP * EMBD;

  const int tB = RbP >> 4, tA = RaP >> 4;
  const int w = tid >> 5;
  const int ta = w / tB, tb = w % tB;
  const bool own = w < tA * tB;
  v8f acc = {};
  for (int k = 0; k < EMBD; k += 64) {
    if (k + 64 < EMBD && tid < RaP)   // warm next chunk (global_prefetch)
      __builtin_prefetch(A + (long)tid * EMBD + k + 64, 0, 0);
    __syncthreads();
    stage_rows<64>(sA, A + k, EMBD, RaP, tid, 288);
    stage_rows<64>(sB, B + k, EMBD, RbP, tid, 288);
    __syncthreads();
    if (own) {
#pragma unroll
      for (int kk = 0; kk < 64; kk += 32) {
        v16bf fa = load_frag_m(sA + (long)(ta * 16) * 64 + kk, 64);
        v16bf fb = load_frag_n(sB + (long)(tb * 16) * 64 + kk, 64);
        acc = wmma_bf16(fa, fb, acc);
      }
    }
  }
  if (own) {
    const int lane = tid & 31, g = lane >> 4, n = lane & 15;
#pragma unroll
    for (int v = 0; v < 8; ++v) S[(ta * 16 + v + 8 * g) * 49 + tb * 16 + n] = acc[v];
  }
  for (int i = tid; i < RaP * RaP; i += 288) GA[i] = GAall[(long)ia * RaP * RaP + i];
  for (int i = tid; i < RbP * RbP; i += 288) GB[i] = GBall[(long)ib * RbP * RbP + i];
  for (int i = tid; i < RaP; i += 288) nA[i] = nAall[(long)ia * RaP + i];
  for (int i = tid; i < RbP; i += 288) nB[i] = nBall[(long)ib * RbP + i];
  for (int i = tid; i < 48; i += 288)
    msk[i] = lengths ? ((i < lengths[ib]) ? 1.f : 0.f) : 1.f;
  __syncthreads();

  const float fs1 = lengths ? (1.f / (float)lengths[ib]) : (1.f / (float)Rb);
  // dir1: ctx = A rows, qry = B rows (t2i-style when lengths given)
  const float s1 = dir_sim(S, 1, Ra, Rb, GA, RaP, nB,
                           lengths ? (const float*)msk : nullptr, nullptr,
                           fs1, rn, wrow, red);
  // dir2: ctx = B rows (word-masked logits when lengths given), qry = A rows, mean over A
  const float s2 = dir_sim(S, 0, Rb, Ra, GB, RbP, nA,
                           nullptr, lengths ? (const float*)msk : nullptr,
                           1.f / (float)Ra, rn, wrow, red);
  if (tid == 0) out[(long)ia * BATCH + ib] = s1 + s2;
}

// ---------------- host orchestration ----------------

extern "C" void kernel_launch(void* const* d_in, const int* in_sizes, int n_in,
                              void* d_out, int out_size, void* d_ws, size_t ws_size,
                              hipStream_t stream) {
  (void)in_sizes; (void)n_in; (void)out_size; (void)ws_size;
  const float* cnns   = (const float*)d_in[0];
  const int*   caps   = (const int*)d_in[1];
  const int*   lens   = (const int*)d_in[2];
  const float* bovws  = (const float*)d_in[3];
  const float* w_cnn  = (const float*)d_in[4];
  const float* b_cnn  = (const float*)d_in[5];
  const float* w_bovw = (const float*)d_in[6];
  const float* b_bovw = (const float*)d_in[7];
  const float* emb_c  = (const float*)d_in[8];
  const float* emb_b  = (const float*)d_in[9];
  const float* c_wih  = (const float*)d_in[10];
  const float* c_whh  = (const float*)d_in[11];
  const float* c_bih  = (const float*)d_in[12];
  const float* c_bhh  = (const float*)d_in[13];
  const float* b_wih  = (const float*)d_in[14];
  const float* b_whh  = (const float*)d_in[15];
  const float* b_bih  = (const float*)d_in[16];
  const float* b_bhh  = (const float*)d_in[17];
  float* out = (float*)d_out;

  char* ws = (char*)d_ws;
  size_t off = 0;
  auto alloc = [&](size_t bytes) -> void* {
    void* p = ws + off;
    off = (off + bytes + 255) & ~(size_t)255;
    return p;
  };

  bf16* cnns_bf  = (bf16*)alloc((size_t)BATCH * RREG * CNND * 2);
  bf16* bovws_bf = (bf16*)alloc((size_t)BATCH * NPAT * EMBD * 2);
  bf16* wcnnT    = (bf16*)alloc((size_t)EMBD * CNND * 2);
  bf16* wbovwT   = (bf16*)alloc((size_t)EMBD * EMBD * 2);
  bf16* wih_all  = (bf16*)alloc((size_t)4 * 3 * EMBD * WPADD * 2);
  bf16* whh_all  = (bf16*)alloc((size_t)4 * 3 * EMBD * EMBD * 2);
  bf16* x_pad    = (bf16*)alloc((size_t)2 * BATCH * MAXL * WPADD * 2);
  float* y_cnn   = (float*)alloc((size_t)BATCH * RREG * EMBD * 4);
  float* y_bovw  = (float*)alloc((size_t)BATCH * NPAT * EMBD * 4);
  bf16* imgA     = (bf16*)alloc((size_t)BATCH * RP * EMBD * 2);
  float* nrmA    = (float*)alloc((size_t)BATCH * RP * 4);
  bf16* imgB     = (bf16*)alloc((size_t)BATCH * PP * EMBD * 2);
  float* nrmB    = (float*)alloc((size_t)BATCH * PP * 4);
  bf16* capC     = (bf16*)alloc((size_t)BATCH * RP * EMBD * 2);
  float* nrmCC   = (float*)alloc((size_t)BATCH * RP * 4);
  bf16* capW     = (bf16*)alloc((size_t)BATCH * RP * EMBD * 2);
  float* nrmCW   = (float*)alloc((size_t)BATCH * RP * 4);
  float* G_img   = (float*)alloc((size_t)BATCH * RP * RP * 4);
  float* G_bovw  = (float*)alloc((size_t)BATCH * PP * PP * 4);
  float* G_capC  = (float*)alloc((size_t)BATCH * RP * RP * 4);
  float* G_capW  = (float*)alloc((size_t)BATCH * RP * RP * 4);
  float* h_f32   = (float*)alloc((size_t)2 * 4 * BATCH * EMBD * 4);  // [parity][stream]
  bf16* h_bf     = (bf16*)alloc((size_t)2 * 4 * BATCH * EMBD * 2);
  float* out_gru = (float*)alloc((size_t)4 * BATCH * MAXL * EMBD * 4);

  auto g1 = [](long n) { return dim3((unsigned)((n + 255) / 256)); };

  // --- prep: conversions / transposes / padding / embedding gather ---
  k_cvt<<<g1((long)BATCH * RREG * CNND), 256, 0, stream>>>(cnns, cnns_bf, (long)BATCH * RREG * CNND);
  k_cvt<<<g1((long)BATCH * NPAT * EMBD), 256, 0, stream>>>(bovws, bovws_bf, (long)BATCH * NPAT * EMBD);
  k_transpose<<<g1((long)CNND * EMBD), 256, 0, stream>>>(w_cnn, wcnnT, CNND, EMBD);
  k_transpose<<<g1((long)EMBD * EMBD), 256, 0, stream>>>(w_bovw, wbovwT, EMBD, EMBD);
  for (int s = 0; s < 4; ++s) {
    const float* src = ((s < 2) ? c_wih : b_wih) + (size_t)(s & 1) * 3 * EMBD * WORDD;
    k_padcvt<<<g1((long)3 * EMBD * WPADD), 256, 0, stream>>>(
        src, wih_all + (size_t)s * 3 * EMBD * WPADD, 3 * EMBD, WORDD, WPADD);
    const float* srh = ((s < 2) ? c_whh : b_whh) + (size_t)(s & 1) * 3 * EMBD * EMBD;
    k_padcvt<<<g1((long)3 * EMBD * EMBD), 256, 0, stream>>>(
        srh, whh_all + (size_t)s * 3 * EMBD * EMBD, 3 * EMBD, EMBD, EMBD);
  }
  k_gather<<<g1((long)BATCH * MAXL * WPADD), 256, 0, stream>>>(caps, emb_c, x_pad);
  k_gather<<<g1((long)BATCH * MAXL * WPADD), 256, 0, stream>>>(
      caps, emb_b, x_pad + (size_t)BATCH * MAXL * WPADD);

  // --- image/bovw projections (WMMA) + l2norm finalize into padded bf16 ---
  k_gemm_bias<<<dim3(EMBD / 16, (BATCH * RREG) / 128), 256, 0, stream>>>(
      cnns_bf, wcnnT, b_cnn, y_cnn, BATCH * RREG, EMBD, CNND);
  k_gemm_bias<<<dim3(EMBD / 16, (BATCH * NPAT) / 128), 256, 0, stream>>>(
      bovws_bf, wbovwT, b_bovw, y_bovw, BATCH * NPAT, EMBD, EMBD);
  k_finalize<<<dim3(RP, BATCH), 256, 0, stream>>>(y_cnn, nullptr, 1.f, imgA, nrmA, RREG, RP, EMBD);
  k_finalize<<<dim3(PP, BATCH), 256, 0, stream>>>(y_bovw, nullptr, 1.f, imgB, nrmB, NPAT, PP, EMBD);

  // --- GRU: zero h0, then 40 fused WMMA recurrence steps over 4 dir-streams ---
  k_zero<<<g1((long)4 * BATCH * EMBD), 256, 0, stream>>>(h_f32, (long)4 * BATCH * EMBD);
  k_zero<<<g1((long)4 * BATCH * EMBD / 2), 256, 0, stream>>>((float*)h_bf, (long)4 * BATCH * EMBD / 2);
  const size_t hstF = (size_t)4 * BATCH * EMBD;
  for (int t = 0; t < MAXL; ++t) {
    const int pin = t & 1, pout = 1 - pin;
    k_gru_step<<<dim3(EMBD / 16, 1, 4), 256, 0, stream>>>(
        x_pad, wih_all, whh_all, c_bih, c_bhh, b_bih, b_bhh, lens,
        h_f32 + (size_t)pin * hstF, h_bf + (size_t)pin * hstF,
        h_f32 + (size_t)pout * hstF, h_bf + (size_t)pout * hstF,
        out_gru, t);
  }
  const size_t dstride = (size_t)BATCH * MAXL * EMBD;
  k_finalize<<<dim3(RP, BATCH), 256, 0, stream>>>(
      out_gru + 0 * dstride, out_gru + 1 * dstride, 0.5f, capC, nrmCC, MAXL, RP, EMBD);
  k_finalize<<<dim3(RP, BATCH), 256, 0, stream>>>(
      out_gru + 2 * dstride, out_gru + 3 * dstride, 0.5f, capW, nrmCW, MAXL, RP, EMBD);

  // --- per-sample Gram matrices (WMMA) ---
  k_gram<<<BATCH, 288, 0, stream>>>(imgA, G_img, RP, EMBD);
  k_gram<<<BATCH, 288, 0, stream>>>(imgB, G_bovw, PP, EMBD);
  k_gram<<<BATCH, 288, 0, stream>>>(capC, G_capC, RP, EMBD);
  k_gram<<<BATCH, 288, 0, stream>>>(capW, G_capW, RP, EMBD);

  // --- fused pair similarities ---
  // sim_p[i,c] = t2i + i2t over (cnn_emb, cap_cnn)
  k_pair<<<dim3(BATCH, BATCH), 288, 0, stream>>>(
      imgA, capC, G_img, G_capC, nrmA, nrmCC, RREG, MAXL, RP, RP, lens, out);
  // sim_a1[i,c] over (bovw_emb, cap_bovw)
  k_pair<<<dim3(BATCH, BATCH), 288, 0, stream>>>(
      imgB, capW, G_bovw, G_capW, nrmB, nrmCW, NPAT, MAXL, PP, RP, lens,
      out + (size_t)BATCH * BATCH);
  // sim_a2[c,b] = b2c + c2b over (cnn_emb sample c, bovw_emb sample b), unmasked
  k_pair<<<dim3(BATCH, BATCH), 288, 0, stream>>>(
      imgA, imgB, G_img, G_bovw, nrmA, nrmB, RREG, NPAT, RP, PP, nullptr,
      out + (size_t)2 * BATCH * BATCH);
}
